// Tokenizer_66322884985148
// MI455X (gfx1250) — matrix operand order repaired and checked by reading, not compiled, not run on hardware
//
#include <hip/hip_runtime.h>
#include <hip/hip_bf16.h>

// ---------------- types ----------------
typedef __attribute__((ext_vector_type(16))) __bf16 v16bf;
typedef __attribute__((ext_vector_type(8)))  __bf16 v8bf;
typedef __attribute__((ext_vector_type(2)))  __bf16 v2bf;
typedef __attribute__((ext_vector_type(8)))  float  v8f;
typedef __attribute__((ext_vector_type(4)))  float  v4f;

// ---------------- problem constants ----------------
#define VARS     8
#define WIN      16
#define STEP     4
#define EMBED    128
#define BATCH    32
#define SAMPLES  8192
#define T_WIN    2044                 // len(arange(0, 8192-16, 4))
#define IN_LEN   256                  // VARS*WIN*K
#define MROWS    (BATCH * T_WIN)      // 65408
#define MTILES   (MROWS / 16)         // 4088
#define MT_PER_WAVE 2                 // 2 M-tiles per wave -> 2x B reuse
#define HSTRIDE  260                  // 256 + 4 pad dwords -> conflict-free

// ============================================================
// Kernel 1: fuzzify  x -> flat (bf16, exact small integers)
// ============================================================
__global__ __launch_bounds__(256) void tok_fuzz(
    const float* __restrict__ x, unsigned int* __restrict__ flat_pairs)
{
    const unsigned n = blockIdx.x * 256u + threadIdx.x;   // exact: 8,372,224 threads
    const int w  = n & 15;
    const int v  = (n >> 4) & 7;
    const int bt = n >> 7;
    const int b  = bt / T_WIN;
    const int t  = bt - b * T_WIN;

    const float d = x[(size_t)(b * VARS + v) * SAMPLES + t * STEP + w];
    const float p = d * 15.0f;
    int ci = (int)floorf(p);
    ci = ci < 0 ? 0 : (ci > 14 ? 14 : ci);
    const float frac = p - (float)ci;

    int t1, t2;
    if (frac > 0.5f)      { t1 = ci + 1; t2 = ci;     }
    else if (frac > 0.0f) { t1 = ci;     t2 = ci + 1; }
    else                  { t1 = ci;     t2 = (ci == 0) ? 1 : 0; }
    v2bf pr;
    pr[0] = (__bf16)(float)t1;
    pr[1] = (__bf16)(float)t2;
    flat_pairs[n] = __builtin_bit_cast(unsigned int, pr);
}

// ============================================================
// Kernel 2: split fp32 weights into bf16 hi + bf16 lo residual
// ============================================================
__global__ __launch_bounds__(256) void tok_prep(
    const float* __restrict__ W1, const float* __restrict__ W2,
    unsigned short* __restrict__ w1hi, unsigned short* __restrict__ w1lo,
    unsigned short* __restrict__ w2hi, unsigned short* __restrict__ w2lo)
{
    const int n = blockIdx.x * 256 + threadIdx.x;         // 98304 threads exact
    if (n < IN_LEN * IN_LEN) {
        const float w = W1[n];
        const __bf16 h = (__bf16)w;
        const __bf16 l = (__bf16)(w - (float)h);
        ((__bf16*)w1hi)[n] = h;
        ((__bf16*)w1lo)[n] = l;
    } else {
        const int m = n - IN_LEN * IN_LEN;
        const float w = W2[m];
        const __bf16 h = (__bf16)w;
        const __bf16 l = (__bf16)(w - (float)h);
        ((__bf16*)w2hi)[m] = h;
        ((__bf16*)w2lo)[m] = l;
    }
}

// ============================================================
// Kernel 3: fused GEMM1 + GEMM2 + LayerNorm, 2 M-tiles per wave.
// ============================================================
__device__ __forceinline__ v8f wmma_bf16(v16bf a, v16bf b, v8f c) {
    return __builtin_amdgcn_wmma_f32_16x16x32_bf16(
        false, a, false, b, (short)0, c, false, false);
}

// Build split-bf16 A fragment (hi, lo) for GEMM2 from 16 f32 read out of LDS.
__device__ __forceinline__ void build_frag(
    const float* __restrict__ hRow, int k0, int half, v16bf& ahi, v16bf& alo)
{
    const v4f f0 = *(const v4f*)(hRow + k0 + half * 16);
    const v4f f1 = *(const v4f*)(hRow + k0 + half * 16 + 4);
    const v4f f2 = *(const v4f*)(hRow + k0 + half * 16 + 8);
    const v4f f3 = *(const v4f*)(hRow + k0 + half * 16 + 12);
#pragma unroll
    for (int i = 0; i < 4; ++i) {
        float f; __bf16 h;
        f = f0[i]; h = (__bf16)f; ahi[i]      = h; alo[i]      = (__bf16)(f - (float)h);
        f = f1[i]; h = (__bf16)f; ahi[4 + i]  = h; alo[4 + i]  = (__bf16)(f - (float)h);
        f = f2[i]; h = (__bf16)f; ahi[8 + i]  = h; alo[8 + i]  = (__bf16)(f - (float)h);
        f = f3[i]; h = (__bf16)f; ahi[12 + i] = h; alo[12 + i] = (__bf16)(f - (float)h);
    }
}

__global__ __launch_bounds__(32) void tok_gemm(
    const unsigned short* __restrict__ flat_,
    const unsigned short* __restrict__ w1hi_, const unsigned short* __restrict__ w1lo_,
    const unsigned short* __restrict__ w2hi_, const unsigned short* __restrict__ w2lo_,
    const float* __restrict__ b1, const float* __restrict__ b2,
    const float* __restrict__ gamma, const float* __restrict__ beta,
    float* __restrict__ out)
{
    __shared__ float hbuf[MT_PER_WAVE][16 * HSTRIDE];     // 33,280 B

    const __bf16* flat = (const __bf16*)flat_;
    const __bf16* w1hi = (const __bf16*)w1hi_;
    const __bf16* w1lo = (const __bf16*)w1lo_;
    const __bf16* w2hi = (const __bf16*)w2hi_;
    const __bf16* w2lo = (const __bf16*)w2lo_;

    const int lane  = threadIdx.x & 31;
    const int half  = lane >> 4;                          // 0|1
    const int l16   = lane & 15;
    const int mt0   = blockIdx.x * MT_PER_WAVE;           // grid covers exactly

    // ---- preload ALL A fragments for both M-tiles (exact in bf16) ----
    // A frag (bf16 16x32): half 0 -> K {k0..+15}; half 1 -> {+16..+31}
    v16bf afr[MT_PER_WAVE][8];                            // 128 VGPRs, resident
#pragma unroll
    for (int mt = 0; mt < MT_PER_WAVE; ++mt) {
        const __bf16* aBase = flat + (size_t)((mt0 + mt) * 16 + l16) * IN_LEN;
#pragma unroll
        for (int kk = 0; kk < 8; ++kk) {
            const v8bf a0 = *(const v8bf*)(aBase + kk * 32 + half * 16);
            const v8bf a1 = *(const v8bf*)(aBase + kk * 32 + 8 + half * 16);
#pragma unroll
            for (int i = 0; i < 8; ++i) { afr[mt][kk][i] = a0[i]; afr[mt][kk][8 + i] = a1[i]; }
        }
    }

    // ---------------- GEMM1: h = flat @ W1^T + b1 -> LDS ----------------
    // Each B fragment (hi or lo) now feeds 2 WMMAs (one per M-tile).
    for (int nt = 0; nt < 16; ++nt) {
        v8f acc0 = {0.f,0.f,0.f,0.f,0.f,0.f,0.f,0.f};
        v8f acc1 = {0.f,0.f,0.f,0.f,0.f,0.f,0.f,0.f};
        const int n = nt * 16 + l16;                      // output column o
        const __bf16* bH = w1hi + (size_t)n * IN_LEN + half * 16;
        const __bf16* bL = w1lo + (size_t)n * IN_LEN + half * 16;
#pragma unroll
        for (int kk = 0; kk < 8; ++kk) {
            const v16bf bh = *(const v16bf*)(bH + kk * 32);
            const v16bf bl = *(const v16bf*)(bL + kk * 32);
            acc0 = wmma_bf16(afr[0][kk], bh, acc0);
            acc0 = wmma_bf16(afr[0][kk], bl, acc0);
            acc1 = wmma_bf16(afr[1][kk], bh, acc1);
            acc1 = wmma_bf16(afr[1][kk], bl, acc1);
        }
        const float bias = b1[n];
#pragma unroll
        for (int vv = 0; vv < 8; ++vv) {                  // D: VGPR v -> row v+8*half
            hbuf[0][(vv + half * 8) * HSTRIDE + n] = acc0[vv] + bias;
            hbuf[1][(vv + half * 8) * HSTRIDE + n] = acc1[vv] + bias;
        }
    }

    // ---------------- GEMM2: z = h @ W2^T + b2 (regs, both tiles) ----------------
    v8f z[MT_PER_WAVE][8];                                // 128 VGPRs
#pragma unroll
    for (int mt = 0; mt < MT_PER_WAVE; ++mt)
#pragma unroll
        for (int nt = 0; nt < 8; ++nt) z[mt][nt] = (v8f){0.f,0.f,0.f,0.f,0.f,0.f,0.f,0.f};

    const float* hRow0 = &hbuf[0][l16 * HSTRIDE];         // A row m = l16
    const float* hRow1 = &hbuf[1][l16 * HSTRIDE];
#pragma unroll 1
    for (int kk = 0; kk < 8; ++kk) {
        const int k0 = kk * 32;
        v16bf ahi0, alo0, ahi1, alo1;
        build_frag(hRow0, k0, half, ahi0, alo0);
        build_frag(hRow1, k0, half, ahi1, alo1);
#pragma unroll
        for (int nt = 0; nt < 8; ++nt) {
            const int n = nt * 16 + l16;
            const v16bf bh = *(const v16bf*)(w2hi + (size_t)n * IN_LEN + half * 16 + k0);
            const v16bf bl = *(const v16bf*)(w2lo + (size_t)n * IN_LEN + half * 16 + k0);
            z[0][nt] = wmma_bf16(ahi0, bh, z[0][nt]);     // 3-term split product
            z[0][nt] = wmma_bf16(alo0, bh, z[0][nt]);
            z[0][nt] = wmma_bf16(ahi0, bl, z[0][nt]);
            z[1][nt] = wmma_bf16(ahi1, bh, z[1][nt]);
            z[1][nt] = wmma_bf16(alo1, bh, z[1][nt]);
            z[1][nt] = wmma_bf16(ahi1, bl, z[1][nt]);
        }
    }

    // ---------------- bias + LayerNorm over EMBED=128, both tiles ----------------
#pragma unroll
    for (int mt = 0; mt < MT_PER_WAVE; ++mt) {
#pragma unroll
        for (int nt = 0; nt < 8; ++nt) {
            const float bb = b2[nt * 16 + l16];
#pragma unroll
            for (int vv = 0; vv < 8; ++vv) z[mt][nt][vv] += bb;
        }

        // Row m's 128 values live in VGPR (m%8) across one 16-lane half.
        float mean[8], rstd[8];
#pragma unroll
        for (int vv = 0; vv < 8; ++vv) {
            float s = 0.f;
#pragma unroll
            for (int nt = 0; nt < 8; ++nt) s += z[mt][nt][vv];
            s += __shfl_xor(s, 1);  s += __shfl_xor(s, 2);
            s += __shfl_xor(s, 4);  s += __shfl_xor(s, 8);  // halves stay separate
            mean[vv] = s * (1.0f / EMBED);
        }
#pragma unroll
        for (int vv = 0; vv < 8; ++vv) {
            float s = 0.f;
#pragma unroll
            for (int nt = 0; nt < 8; ++nt) { const float d = z[mt][nt][vv] - mean[vv]; s += d * d; }
            s += __shfl_xor(s, 1);  s += __shfl_xor(s, 2);
            s += __shfl_xor(s, 4);  s += __shfl_xor(s, 8);
            rstd[vv] = rsqrtf(s * (1.0f / EMBED) + 1e-5f);
        }

#pragma unroll
        for (int nt = 0; nt < 8; ++nt) {
            const int o = nt * 16 + l16;
            const float g = gamma[o], be = beta[o];
#pragma unroll
            for (int vv = 0; vv < 8; ++vv) {
                const int m = vv + half * 8;
                out[(size_t)((mt0 + mt) * 16 + m) * EMBED + o] =
                    (z[mt][nt][vv] - mean[vv]) * rstd[vv] * g + be;
            }
        }
    }
}

// ============================================================
// launcher
// ============================================================
extern "C" void kernel_launch(void* const* d_in, const int* in_sizes, int n_in,
                              void* d_out, int out_size, void* d_ws, size_t ws_size,
                              hipStream_t stream)
{
    const float* x     = (const float*)d_in[0];
    const float* W1    = (const float*)d_in[1];
    const float* b1    = (const float*)d_in[2];
    const float* W2    = (const float*)d_in[3];
    const float* b2    = (const float*)d_in[4];
    const float* gamma = (const float*)d_in[5];
    const float* beta  = (const float*)d_in[6];

    char* ws = (char*)d_ws;
    size_t off = 0;
    unsigned short* flat = (unsigned short*)(ws + off);   // 65408*256 bf16 = 33.5 MB
    off += (size_t)MROWS * IN_LEN * 2;  off = (off + 255) & ~(size_t)255;
    unsigned short* w1hi = (unsigned short*)(ws + off); off += IN_LEN * IN_LEN * 2;
    unsigned short* w1lo = (unsigned short*)(ws + off); off += IN_LEN * IN_LEN * 2;
    unsigned short* w2hi = (unsigned short*)(ws + off); off += EMBED * IN_LEN * 2;
    unsigned short* w2lo = (unsigned short*)(ws + off); off += EMBED * IN_LEN * 2;

    // 1) fuzzify: 32*2044*8*16 = 8,372,224 threads (exact multiple of 256)
    tok_fuzz<<<(MROWS * VARS * WIN) / 256, 256, 0, stream>>>(x, (unsigned int*)flat);

    // 2) weight split: 65536 + 32768 = 98304 threads
    tok_prep<<<(IN_LEN * IN_LEN + EMBED * IN_LEN) / 256, 256, 0, stream>>>(
        W1, W2, w1hi, w1lo, w2hi, w2lo);

    // 3) fused GEMM+GEMM+LayerNorm: 4088 M-tiles, 2 per wave, 1 wave/block
    tok_gemm<<<MTILES / MT_PER_WAVE, 32, 0, stream>>>(
        flat, w1hi, w1lo, w2hi, w2lo, b1, b2, gamma, beta, (float*)d_out);
}
